// FastWeightMLP_10325101379646
// MI455X (gfx1250) — compile-verified, matrix operand order checked by reading, not run on hardware
//
#include <hip/hip_runtime.h>
#include <hip/hip_bf16.h>

// ---------------------------------------------------------------------------
// Fast-weight SwiGLU MLP trained online over 64 sequential 64-step chunks.
// Single persistent kernel: 128 blocks x 128 threads, software grid barriers
// between the 5 dependent phases of each chunk step. All matrix math via
// v_wmma_f32_16x16x32_bf16 (wave32 CDNA5 WMMA); f32 master weights, bf16
// GEMM operand shadows. Phases are __noinline__ device functions (per-phase
// register allocation) and all pointers are address_space(1) so the phases
// emit global_load/global_store (LOADcnt only) instead of FLAT ops that
// would tie up DScnt and the LDS arbitration path.
// ---------------------------------------------------------------------------

typedef __attribute__((ext_vector_type(16))) __bf16 v16bf;
typedef __attribute__((ext_vector_type(8)))  __bf16 v8bf;
typedef __attribute__((ext_vector_type(8)))  float  v8f;

#define GLB __attribute__((address_space(1)))
typedef GLB __bf16 gbf16;
typedef GLB float  gfloat;
typedef GLB v8bf   gv8bf;

__device__ __forceinline__ const gbf16* gptr(const __bf16* p) {
  return (const gbf16*)(unsigned long long)p;
}
__device__ __forceinline__ gbf16* gptr(__bf16* p) {
  return (gbf16*)(unsigned long long)p;
}
__device__ __forceinline__ const gfloat* gptr(const float* p) {
  return (const gfloat*)(unsigned long long)p;
}
__device__ __forceinline__ gfloat* gptr(float* p) {
  return (gfloat*)(unsigned long long)p;
}

namespace {
constexpr int kB    = 8;
constexpr int kT    = 4096;
constexpr int kDin  = 512;
constexpr int kDout = 512;
constexpr int kS    = 1024;
constexpr int kCH   = 64;
constexpr int kM    = kB * kCH;   // 512 rows per chunk step
constexpr int kNCH  = kT / kCH;   // 64 sequential chunks
constexpr int kNBLK = 128;        // persistent grid: must all be co-resident
constexpr int kBT   = 128;        // threads per block (4 waves)
constexpr float kLR = 0.1f;
constexpr float kGradScale = 2.0f / (float)(kM * kDout);
}

#define PHASE __device__ __attribute__((noinline)) void

// --- WMMA fragment load -----------------------------------------------------
// bf16 16x32 A / 32x16 B operand: lane l owns row/col (l&15); VGPR v holds the
// K pair at (v<4?0:16) + (l>>4)*8 + (v&3)*2  ==> two contiguous 16-byte global
// loads per lane at K offsets {half*8, 16+half*8} from a row-major [rows,K]
// matrix. Identical pattern serves A tiles and "W[N,K]^T-as-B" tiles.
__device__ __forceinline__ v16bf load_frag(const gbf16* __restrict__ base,
                                           int row, int ldk, int k0, int lane) {
  const gbf16* p = base + (size_t)row * ldk + k0 + ((lane >> 4) << 3);
  v8bf lo = *(const gv8bf*)(p);
  v8bf hi = *(const gv8bf*)(p + 16);
  return __builtin_shufflevector(lo, hi, 0, 1, 2, 3, 4, 5, 6, 7,
                                 8, 9, 10, 11, 12, 13, 14, 15);
}

__device__ __forceinline__ v8f wmma_bf16(v16bf a, v16bf b, v8f c) {
  return __builtin_amdgcn_wmma_f32_16x16x32_bf16(false, a, false, b,
                                                 (short)0, c, false, false);
}

// --- 32x32-per-wave GEMM main loops: C[M,N] = A[M,K] x W[N,K]^T -------------
template <int K>
__device__ __forceinline__ void gemm_tile(const gbf16* __restrict__ A,
                                          const gbf16* __restrict__ W,
                                          int m0, int n0, int lane,
                                          v8f acc[2][2]) {
  const int r = lane & 15;
#pragma unroll 4
  for (int k = 0; k < K; k += 32) {
    v16bf a0 = load_frag(A, m0 + r,      K, k, lane);
    v16bf a1 = load_frag(A, m0 + 16 + r, K, k, lane);
    v16bf b0 = load_frag(W, n0 + r,      K, k, lane);
    v16bf b1 = load_frag(W, n0 + 16 + r, K, k, lane);
    acc[0][0] = wmma_bf16(a0, b0, acc[0][0]);
    acc[0][1] = wmma_bf16(a0, b1, acc[0][1]);
    acc[1][0] = wmma_bf16(a1, b0, acc[1][0]);
    acc[1][1] = wmma_bf16(a1, b1, acc[1][1]);
  }
}

// Dual-B variant (shared A fragments, two weight matrices).
template <int K>
__device__ __forceinline__ void gemm_tile2(const gbf16* __restrict__ A,
                                           const gbf16* __restrict__ W1,
                                           const gbf16* __restrict__ W2,
                                           int m0, int n0, int lane,
                                           v8f acc1[2][2], v8f acc2[2][2]) {
  const int r = lane & 15;
#pragma unroll 2
  for (int k = 0; k < K; k += 32) {
    v16bf a0 = load_frag(A, m0 + r,      K, k, lane);
    v16bf a1 = load_frag(A, m0 + 16 + r, K, k, lane);
    v16bf u0 = load_frag(W1, n0 + r,      K, k, lane);
    v16bf u1 = load_frag(W1, n0 + 16 + r, K, k, lane);
    v16bf g0 = load_frag(W2, n0 + r,      K, k, lane);
    v16bf g1 = load_frag(W2, n0 + 16 + r, K, k, lane);
    acc1[0][0] = wmma_bf16(a0, u0, acc1[0][0]);
    acc1[0][1] = wmma_bf16(a0, u1, acc1[0][1]);
    acc1[1][0] = wmma_bf16(a1, u0, acc1[1][0]);
    acc1[1][1] = wmma_bf16(a1, u1, acc1[1][1]);
    acc2[0][0] = wmma_bf16(a0, g0, acc2[0][0]);
    acc2[0][1] = wmma_bf16(a0, g1, acc2[0][1]);
    acc2[1][0] = wmma_bf16(a1, g0, acc2[1][0]);
    acc2[1][1] = wmma_bf16(a1, g1, acc2[1][1]);
  }
}

// C/D element owned by (vgpr rr, lane l): m_local = rr + 8*(l>>4), n_local=l&15
#define EPILOGUE_FOR(lane)                                                    \
  const int eh = (lane) >> 4, en = (lane) & 15;                               \
  for (int mt = 0; mt < 2; ++mt)                                              \
    for (int nt = 0; nt < 2; ++nt)                                            \
      for (int rr = 0; rr < 8; ++rr)

// --- phases (noinline: per-phase register allocation) -----------------------

PHASE p_prep(const gfloat* __restrict__ q, gbf16* Qc, gbf16* QcT, int c,
             int gtid) {
  for (int idx = gtid; idx < kM * kDin; idx += kNBLK * kBT) {
    int m = idx >> 9, k = idx & (kDin - 1);
    int b = m >> 6, tt = m & (kCH - 1);
    float v = q[((size_t)(b * kT + c * kCH + tt)) * kDin + k];
    __bf16 vb = (__bf16)v;
    Qc[(size_t)m * kDin + k] = vb;
    QcT[(size_t)k * kM + m]  = vb;
  }
}

PHASE p_fwd_upgate(const gbf16* Qc, const gbf16* Wup, const gbf16* Wg,
                   const gfloat* bup, const gfloat* bg, gfloat* upf,
                   gfloat* gatef, gfloat* zgf, gbf16* Hb, gbf16* HTb, int bid,
                   int lane, int wm, int wn) {
  int m0 = (bid >> 4) * 64 + wm;           // [512,1024]: 8x16 = 128 tiles
  int n0 = (bid & 15) * 64 + wn;
  v8f au[2][2] = {}, ag[2][2] = {};
  gemm_tile2<kDin>(Qc, Wup, Wg, m0, n0, lane, au, ag);
  EPILOGUE_FOR(lane) {
    int m = m0 + mt * 16 + rr + eh * 8;
    int n = n0 + nt * 16 + en;
    float zu  = au[mt][nt][rr] + bup[n];
    float zg  = ag[mt][nt][rr] + bg[n];
    float sig = 1.0f / (1.0f + __expf(-zg));
    float gt  = zg * sig;
    float h   = zu * gt;
    size_t idx = (size_t)m * kS + n;
    upf[idx] = zu; gatef[idx] = gt; zgf[idx] = zg;
    __bf16 hb = (__bf16)h;
    Hb[idx] = hb;
    HTb[(size_t)n * kM + m] = hb;
  }
}

PHASE p_fwd_down(const gbf16* Hb, const gbf16* Wdown, const gfloat* bdown,
                 const gfloat* __restrict__ ref_v, gfloat* __restrict__ out,
                 int c, gfloat* doutf, gbf16* doutb, gbf16* doutTb, int bid,
                 int lane, int wm, int wn) {
  int m0 = (bid >> 3) * 64 + wm;           // [512,512]: 8x8 = 64 tiles
  int n0 = (bid & 7) * 64 + wn;
  v8f acc[2][2] = {};
  gemm_tile<kS>(Hb, Wdown, m0, n0, lane, acc);
  EPILOGUE_FOR(lane) {
    int m = m0 + mt * 16 + rr + eh * 8;
    int n = n0 + nt * 16 + en;
    int b = m >> 6, tt = m & (kCH - 1);
    size_t gi = ((size_t)(b * kT + c * kCH + tt)) * kDout + n;
    float o = acc[mt][nt][rr] + bdown[n];
    out[gi] = o;
    float dv = kGradScale * (o - ref_v[gi]);
    doutf[(size_t)m * kDout + n] = dv;
    __bf16 db = (__bf16)dv;
    doutb[(size_t)m * kDout + n] = db;
    doutTb[(size_t)n * kM + m]   = db;
  }
}

PHASE p_bwd_dh(const gbf16* doutb, const gbf16* WdownT, const gfloat* upf,
               const gfloat* gatef, const gfloat* zgf, gfloat* dupf,
               gfloat* dzgf, gbf16* dUpT, gbf16* dZgT, int bid, int lane,
               int wm, int wn) {
  int m0 = (bid >> 4) * 64 + wm;           // [512,1024]: 128 tiles
  int n0 = (bid & 15) * 64 + wn;
  v8f acc[2][2] = {};
  gemm_tile<kDout>(doutb, WdownT, m0, n0, lane, acc);
  EPILOGUE_FOR(lane) {
    int m = m0 + mt * 16 + rr + eh * 8;
    int n = n0 + nt * 16 + en;
    size_t idx = (size_t)m * kS + n;
    float dh  = acc[mt][nt][rr];
    float zg  = zgf[idx];
    float sig = 1.0f / (1.0f + __expf(-zg));
    float dup = dh * gatef[idx];
    float dzg = dh * upf[idx] * sig * (1.0f + zg * (1.0f - sig));
    dupf[idx] = dup; dzgf[idx] = dzg;
    dUpT[(size_t)n * kM + m] = (__bf16)dup;
    dZgT[(size_t)n * kM + m] = (__bf16)dzg;
  }
}

PHASE p_grad_wdown(const gbf16* doutTb, const gbf16* HTb, gfloat* wdown_m,
                   gbf16* Wdown_bf, gbf16* WdownT_bf, int bid, int lane,
                   int wm, int wn) {
  int m0 = (bid >> 4) * 64 + wm;           // dWdown [512,1024]: 128 tiles
  int n0 = (bid & 15) * 64 + wn;
  v8f acc[2][2] = {};
  gemm_tile<kM>(doutTb, HTb, m0, n0, lane, acc);
  EPILOGUE_FOR(lane) {
    int d = m0 + mt * 16 + rr + eh * 8;
    int s = n0 + nt * 16 + en;
    float g = fminf(1.0f, fmaxf(-1.0f, acc[mt][nt][rr]));
    size_t idx = (size_t)d * kS + s;
    float w = wdown_m[idx] - kLR * g;
    wdown_m[idx] = w;
    __bf16 wb = (__bf16)w;
    Wdown_bf[idx] = wb;
    WdownT_bf[(size_t)s * kDout + d] = wb;
  }
}

PHASE p_grad_wupgate(const gbf16* dUpT, const gbf16* dZgT, const gbf16* QcT,
                     gfloat* wup_m, gfloat* wgate_m, gbf16* Wup_bf,
                     gbf16* Wgate_bf, int bid, int lane, int wm, int wn) {
  int m0 = (bid >> 3) * 64 + wm;           // [1024,512]: 16x8 = 128 tiles
  int n0 = (bid & 7) * 64 + wn;
  v8f accu[2][2] = {}, accg[2][2] = {};
  const int r = lane & 15;
#pragma unroll 2
  for (int k = 0; k < kM; k += 32) {       // shared-B dual-A
    v16bf b0 = load_frag(QcT,  n0 + r,      kM, k, lane);
    v16bf b1 = load_frag(QcT,  n0 + 16 + r, kM, k, lane);
    v16bf u0 = load_frag(dUpT, m0 + r,      kM, k, lane);
    v16bf u1 = load_frag(dUpT, m0 + 16 + r, kM, k, lane);
    v16bf g0 = load_frag(dZgT, m0 + r,      kM, k, lane);
    v16bf g1 = load_frag(dZgT, m0 + 16 + r, kM, k, lane);
    accu[0][0] = wmma_bf16(u0, b0, accu[0][0]);
    accu[0][1] = wmma_bf16(u0, b1, accu[0][1]);
    accu[1][0] = wmma_bf16(u1, b0, accu[1][0]);
    accu[1][1] = wmma_bf16(u1, b1, accu[1][1]);
    accg[0][0] = wmma_bf16(g0, b0, accg[0][0]);
    accg[0][1] = wmma_bf16(g0, b1, accg[0][1]);
    accg[1][0] = wmma_bf16(g1, b0, accg[1][0]);
    accg[1][1] = wmma_bf16(g1, b1, accg[1][1]);
  }
  EPILOGUE_FOR(lane) {
    int s = m0 + mt * 16 + rr + eh * 8;
    int d = n0 + nt * 16 + en;
    size_t idx = (size_t)s * kDin + d;
    float gu = fminf(1.0f, fmaxf(-1.0f, accu[mt][nt][rr]));
    float gg = fminf(1.0f, fmaxf(-1.0f, accg[mt][nt][rr]));
    float wu = wup_m[idx] - kLR * gu;
    float wg = wgate_m[idx] - kLR * gg;
    wup_m[idx] = wu;   Wup_bf[idx]   = (__bf16)wu;
    wgate_m[idx] = wg; Wgate_bf[idx] = (__bf16)wg;
  }
}

PHASE p_bias(const gfloat* doutf, const gfloat* dupf, const gfloat* dzgf,
             gfloat* bdown_m, gfloat* bup_m, gfloat* bgate_m, int j) {
  if (j < kDout) {
    float s = 0.0f;
    for (int m = 0; m < kM; ++m) s += doutf[(size_t)m * kDout + j];
    bdown_m[j] -= kLR * fminf(1.0f, fmaxf(-1.0f, s));
  } else if (j < kDout + kS) {
    int n = j - kDout; float s = 0.0f;
    for (int m = 0; m < kM; ++m) s += dupf[(size_t)m * kS + n];
    bup_m[n] -= kLR * fminf(1.0f, fmaxf(-1.0f, s));
  } else if (j < kDout + 2 * kS) {
    int n = j - kDout - kS; float s = 0.0f;
    for (int m = 0; m < kM; ++m) s += dzgf[(size_t)m * kS + n];
    bgate_m[n] -= kLR * fminf(1.0f, fmaxf(-1.0f, s));
  }
}

// --- software grid barrier (all kNBLK blocks guaranteed co-resident) --------
__device__ __forceinline__ void gsync(unsigned* bar, unsigned& epoch) {
  epoch += (unsigned)kNBLK;
  __threadfence();                 // publish this block's writes device-wide
  __syncthreads();
  if (threadIdx.x == 0) {
    __hip_atomic_fetch_add(bar, 1u, __ATOMIC_RELEASE, __HIP_MEMORY_SCOPE_AGENT);
    while (__hip_atomic_load(bar, __ATOMIC_ACQUIRE,
                             __HIP_MEMORY_SCOPE_AGENT) < epoch)
      __builtin_amdgcn_s_sleep(2);
  }
  __syncthreads();
  __threadfence();                 // drop stale near-cache lines before reads
}

// --- init: f32 masters + bf16 weight shadows + barrier reset ----------------
__global__ __launch_bounds__(256) void k_init(
    const float* __restrict__ wup_in, const float* __restrict__ wgate_in,
    const float* __restrict__ wdown_in, const float* __restrict__ bup_in,
    const float* __restrict__ bgate_in, const float* __restrict__ bdown_in,
    float* wup_m, float* wgate_m, float* wdown_m,
    float* bup_m, float* bgate_m, float* bdown_m,
    __bf16* Wup_bf, __bf16* Wgate_bf, __bf16* Wdown_bf, __bf16* WdownT_bf,
    unsigned* bar) {
  int i = blockIdx.x * 256 + threadIdx.x;
  for (int idx = i; idx < kS * kDin; idx += gridDim.x * 256) {
    float wu = wup_in[idx];   wup_m[idx] = wu;   Wup_bf[idx]   = (__bf16)wu;
    float wg = wgate_in[idx]; wgate_m[idx] = wg; Wgate_bf[idx] = (__bf16)wg;
    float wd = wdown_in[idx]; wdown_m[idx] = wd; Wdown_bf[idx] = (__bf16)wd;
    int d = idx >> 10, s = idx & (kS - 1);   // wdown is [Dout, S]
    WdownT_bf[(size_t)s * kDout + d] = (__bf16)wd;
  }
  if (i < kS)   { bup_m[i] = bup_in[i]; bgate_m[i] = bgate_in[i]; }
  if (i < kDout) bdown_m[i] = bdown_in[i];
  if (i == 0)    *bar = 0u;
}

// --- persistent training kernel ---------------------------------------------
__global__ __launch_bounds__(kBT)
__attribute__((amdgpu_waves_per_eu(2))) void k_train(
    const float* __restrict__ q, const float* __restrict__ ref_v,
    float* __restrict__ out,
    float* wup_m, float* wgate_m, float* wdown_m,
    float* bup_m, float* bgate_m, float* bdown_m,
    __bf16* Wup_bf, __bf16* Wgate_bf, __bf16* Wdown_bf, __bf16* WdownT_bf,
    __bf16* Qc, __bf16* QcT,
    float* upf, float* gatef, float* zgf, __bf16* Hb, __bf16* HTb,
    float* doutf, __bf16* doutb, __bf16* doutTb,
    float* dupf, float* dzgf, __bf16* dUpT, __bf16* dZgT,
    unsigned* bar) {
  const int tid  = threadIdx.x;
  const int bid  = blockIdx.x;
  const int lane = tid & 31;
  const int wave = tid >> 5;                 // 4 waves: 2x2 of 32x32 tiles
  const int wm   = (wave >> 1) * 32;
  const int wn   = (wave & 1) * 32;
  const int gtid = bid * kBT + tid;
  unsigned epoch = 0;

  for (int c = 0; c < kNCH; ++c) {
    p_prep(gptr(q), gptr(Qc), gptr(QcT), c, gtid);
    gsync(bar, epoch);

    p_fwd_upgate(gptr(Qc), gptr(Wup_bf), gptr(Wgate_bf), gptr(bup_m),
                 gptr(bgate_m), gptr(upf), gptr(gatef), gptr(zgf), gptr(Hb),
                 gptr(HTb), bid, lane, wm, wn);
    gsync(bar, epoch);

    if (bid < 64)
      p_fwd_down(gptr(Hb), gptr(Wdown_bf), gptr(bdown_m), gptr(ref_v),
                 gptr(out), c, gptr(doutf), gptr(doutb), gptr(doutTb), bid,
                 lane, wm, wn);
    gsync(bar, epoch);

    p_bwd_dh(gptr(doutb), gptr(WdownT_bf), gptr(upf), gptr(gatef), gptr(zgf),
             gptr(dupf), gptr(dzgf), gptr(dUpT), gptr(dZgT), bid, lane, wm,
             wn);
    gsync(bar, epoch);

    // independent weight-grad GEMMs + bias updates, all consumed next chunk
    p_grad_wdown(gptr(doutTb), gptr(HTb), gptr(wdown_m), gptr(Wdown_bf),
                 gptr(WdownT_bf), bid, lane, wm, wn);
    p_grad_wupgate(gptr(dUpT), gptr(dZgT), gptr(QcT), gptr(wup_m),
                   gptr(wgate_m), gptr(Wup_bf), gptr(Wgate_bf), bid, lane, wm,
                   wn);
    p_bias(gptr(doutf), gptr(dupf), gptr(dzgf), gptr(bdown_m), gptr(bup_m),
           gptr(bgate_m), gtid);
    gsync(bar, epoch);
  }
}

// ---------------------------------------------------------------------------
extern "C" void kernel_launch(void* const* d_in, const int* in_sizes, int n_in,
                              void* d_out, int out_size, void* d_ws,
                              size_t ws_size, hipStream_t stream) {
  const float* q        = (const float*)d_in[0];
  const float* ref_v    = (const float*)d_in[1];
  const float* wup_in   = (const float*)d_in[2];
  const float* bup_in   = (const float*)d_in[3];
  const float* wgate_in = (const float*)d_in[4];
  const float* bgate_in = (const float*)d_in[5];
  const float* wdown_in = (const float*)d_in[6];
  const float* bdown_in = (const float*)d_in[7];
  float* out = (float*)d_out;

  char* p = (char*)d_ws;
  auto alloc = [&](size_t bytes) -> void* {
    void* r = (void*)p;
    p += (bytes + 255) & ~(size_t)255;
    return r;
  };
  float* wup_m   = (float*)alloc(sizeof(float) * kS * kDin);
  float* wgate_m = (float*)alloc(sizeof(float) * kS * kDin);
  float* wdown_m = (float*)alloc(sizeof(float) * kDout * kS);
  float* bup_m   = (float*)alloc(sizeof(float) * kS);
  float* bgate_m = (float*)alloc(sizeof(float) * kS);
  float* bdown_m = (float*)alloc(sizeof(float) * kDout);
  __bf16* Wup_bf    = (__bf16*)alloc(sizeof(__bf16) * kS * kDin);
  __bf16* Wgate_bf  = (__bf16*)alloc(sizeof(__bf16) * kS * kDin);
  __bf16* Wdown_bf  = (__bf16*)alloc(sizeof(__bf16) * kDout * kS);
  __bf16* WdownT_bf = (__bf16*)alloc(sizeof(__bf16) * kS * kDout);
  __bf16* Qc_bf  = (__bf16*)alloc(sizeof(__bf16) * kM * kDin);
  __bf16* QcT_bf = (__bf16*)alloc(sizeof(__bf16) * kDin * kM);
  float* upf   = (float*)alloc(sizeof(float) * kM * kS);
  float* gatef = (float*)alloc(sizeof(float) * kM * kS);
  float* zgf   = (float*)alloc(sizeof(float) * kM * kS);
  __bf16* Hb   = (__bf16*)alloc(sizeof(__bf16) * kM * kS);
  __bf16* HTb  = (__bf16*)alloc(sizeof(__bf16) * kS * kM);
  float* doutf   = (float*)alloc(sizeof(float) * kM * kDout);
  __bf16* doutb  = (__bf16*)alloc(sizeof(__bf16) * kM * kDout);
  __bf16* doutTb = (__bf16*)alloc(sizeof(__bf16) * kDout * kM);
  float* dupf  = (float*)alloc(sizeof(float) * kM * kS);
  float* dzgf  = (float*)alloc(sizeof(float) * kM * kS);
  __bf16* dUpT = (__bf16*)alloc(sizeof(__bf16) * kS * kM);
  __bf16* dZgT = (__bf16*)alloc(sizeof(__bf16) * kS * kM);
  unsigned* bar = (unsigned*)alloc(sizeof(unsigned) * 64);

  k_init<<<2048, 256, 0, stream>>>(wup_in, wgate_in, wdown_in, bup_in,
                                   bgate_in, bdown_in, wup_m, wgate_m, wdown_m,
                                   bup_m, bgate_m, bdown_m, Wup_bf, Wgate_bf,
                                   Wdown_bf, WdownT_bf, bar);

  k_train<<<kNBLK, kBT, 0, stream>>>(
      q, ref_v, out, wup_m, wgate_m, wdown_m, bup_m, bgate_m, bdown_m,
      Wup_bf, Wgate_bf, Wdown_bf, WdownT_bf, Qc_bf, QcT_bf, upf, gatef, zgf,
      Hb, HTb, doutf, doutb, doutTb, dupf, dzgf, dUpT, dZgT, bar);
}
